// AttentionForBeit_50551765074268
// MI455X (gfx1250) — matrix operand order repaired and checked by reading, not compiled, not run on hardware
//
#include <hip/hip_runtime.h>
#include <hip/hip_bf16.h>

typedef __attribute__((ext_vector_type(16))) __bf16 v16bf;
typedef __attribute__((ext_vector_type(8)))  float  v8f;

#define DIM   768
#define NH    12
#define HD    64
#define NTOK  1024
#define BATCH 8
#define BM    128
#define BN    128
#define BK    32

// ---------- bf16 helpers ----------
__device__ __forceinline__ __bf16 f2bf(float f) {
  unsigned u = __builtin_bit_cast(unsigned, f);
  unsigned r = u + 0x7FFFu + ((u >> 16) & 1u);   // round-to-nearest-even
  return __builtin_bit_cast(__bf16, (unsigned short)(r >> 16));
}
__device__ __forceinline__ float bf2f(__bf16 b) {
  unsigned u = ((unsigned)__builtin_bit_cast(unsigned short, b)) << 16;
  return __builtin_bit_cast(float, u);
}
__device__ __forceinline__ unsigned pack_bf2(float lo, float hi) {
  return ((unsigned)__builtin_bit_cast(unsigned short, f2bf(lo))) |
         (((unsigned)__builtin_bit_cast(unsigned short, f2bf(hi))) << 16);
}

// ---------- WMMA wrapper ----------
__device__ __forceinline__ v8f wmma_bf16(v16bf a, v16bf b, v8f c) {
  return __builtin_amdgcn_wmma_f32_16x16x32_bf16(false, a, false, b, (short)0, c,
                                                 false, false);
}

// ---------- fragment loaders (ISA 7.12.2 layouts) ----------
// A matrix 16x32 (MxK), accessor base[m*stride + k], pairs contiguous in k.
__device__ __forceinline__ v16bf load_a_frag(const __bf16* base, int stride) {
  int lane = threadIdx.x & 31;
  int hi = lane >> 4;          // lanes 16-31: K offset +8
  int m  = lane & 15;
  v16bf a;
#pragma unroll
  for (int p = 0; p < 8; ++p) {
    int k0 = ((p & 3) * 2) + ((p >= 4) ? 16 : 0) + (hi ? 8 : 0);
    unsigned u = *(const unsigned*)(base + m * stride + k0);
    a[2 * p]     = __builtin_bit_cast(__bf16, (unsigned short)(u & 0xFFFFu));
    a[2 * p + 1] = __builtin_bit_cast(__bf16, (unsigned short)(u >> 16));
  }
  return a;
}
// B matrix 32x16 (KxN) from [n][k]-major storage: element(k,n)=base[n*stride+k].
__device__ __forceinline__ v16bf load_b_frag_nk(const __bf16* base, int stride) {
  int lane = threadIdx.x & 31;
  int n    = lane & 15;
  int koff = (lane >> 4) ? 8 : 0;    // VGPR v holds K=2(v&3)+{0,1} (+16 for v>=4, +8 for hi half)
  v16bf b;
#pragma unroll
  for (int p = 0; p < 8; ++p) {
    unsigned u = *(const unsigned*)(base + n * stride + koff + (p & 3) * 2 + ((p >= 4) ? 16 : 0));
    b[2 * p]     = __builtin_bit_cast(__bf16, (unsigned short)(u & 0xFFFFu));
    b[2 * p + 1] = __builtin_bit_cast(__bf16, (unsigned short)(u >> 16));
  }
  return b;
}

// ---------- kernel 1: gather relative-position bias -> (nH, N, N) bf16 ----------
__global__ __launch_bounds__(256) void beit_bias_gather(
    const float* __restrict__ table, const int* __restrict__ idx,
    __bf16* __restrict__ biasbuf) {
  int t = blockIdx.x * 256 + threadIdx.x;        // over NH*N*N = 12,582,912
  int h  = t / (NTOK * NTOK);
  int nm = t - h * (NTOK * NTOK);
  biasbuf[t] = f2bf(table[idx[nm] * NH + h]);
}

// ---------- kernel 2: QKV GEMM (8192 x 2304 x 768) ----------
// 256 threads = 8 waves in a 4(M) x 2(N) grid; each wave: 32x64 of a 128x128 tile.
__global__ __launch_bounds__(256) void beit_qkv_gemm(
    const float* __restrict__ x, const float* __restrict__ w,
    const float* __restrict__ qb, const float* __restrict__ vb,
    __bf16* __restrict__ qbuf, __bf16* __restrict__ kbuf,
    __bf16* __restrict__ vbufT) {
  __shared__ unsigned ldsAw[BM][BK / 2];   // 8 KB, packed bf16 pairs
  __shared__ unsigned ldsBw[BN][BK / 2];   // 8 KB, [n][k] packed pairs
  int m0 = blockIdx.x * BM;
  int n0 = blockIdx.y * BN;
  int wave = threadIdx.x >> 5;
  int waveM = wave >> 1, waveN = wave & 1;
  v8f acc[2][4] = {};

  for (int k0 = 0; k0 < DIM; k0 += BK) {
    __syncthreads();
    for (int i = threadIdx.x; i < (BM * BK) / 2; i += 256) {
      int r = i >> 4, c = (i & 15) * 2;
      const float* p = &x[(long)(m0 + r) * DIM + k0 + c];
      ldsAw[r][i & 15] = pack_bf2(p[0], p[1]);
    }
    for (int i = threadIdx.x; i < (BN * BK) / 2; i += 256) {
      int r = i >> 4, c = (i & 15) * 2;
      const float* p = &w[(long)(n0 + r) * DIM + k0 + c];
      ldsBw[r][i & 15] = pack_bf2(p[0], p[1]);
    }
    if (k0 + BK < DIM) {                    // prefetch next K-step tiles into L2/WGP$
      __builtin_prefetch(&x[(long)(m0 + (threadIdx.x & 127)) * DIM + k0 + BK], 0, 1);
      __builtin_prefetch(&w[(long)(n0 + (threadIdx.x & 127)) * DIM + k0 + BK], 0, 1);
    }
    __syncthreads();
    const __bf16* lA = (const __bf16*)ldsAw;
    const __bf16* lB = (const __bf16*)ldsBw;
    v16bf a0 = load_a_frag(lA + (waveM * 32) * BK, BK);
    v16bf a1 = load_a_frag(lA + (waveM * 32 + 16) * BK, BK);
#pragma unroll
    for (int t = 0; t < 4; ++t) {
      v16bf bf = load_b_frag_nk(lB + (waveN * 64 + t * 16) * BK, BK);
      acc[0][t] = wmma_bf16(a0, bf, acc[0][t]);
      acc[1][t] = wmma_bf16(a1, bf, acc[1][t]);
    }
  }

  int lane = threadIdx.x & 31;
  int hi = lane >> 4, nn = lane & 15;
  int sel = n0 / DIM;                       // block-uniform (BN divides DIM)
  int cc0 = n0 - sel * DIM + waveN * 64;
  const float scale = 0.125f;               // 64^-0.5
#pragma unroll
  for (int u = 0; u < 2; ++u)
#pragma unroll
    for (int t = 0; t < 4; ++t)
#pragma unroll
      for (int r = 0; r < 8; ++r) {
        int gm = m0 + waveM * 32 + u * 16 + r + hi * 8;
        int cc = cc0 + t * 16 + nn;
        int h = cc >> 6, d = cc & 63;
        int b = gm >> 10, ntok = gm & 1023;
        float v = acc[u][t][r];
        if (sel == 0)
          qbuf[(((long)(b * NH + h)) * NTOK + ntok) * HD + d] =
              f2bf((v + qb[cc]) * scale);
        else if (sel == 1)
          kbuf[(((long)(b * NH + h)) * NTOK + ntok) * HD + d] = f2bf(v);
        else  // V stored transposed: (b,h,d,ntok) for contiguous PV fragments
          vbufT[(((long)(b * NH + h)) * HD + d) * NTOK + ntok] =
              f2bf(v + vb[cc]);
      }
}

// ---------- kernel 3: fused attention (flash-style, per-wave 16-row Q tile) ----------
__global__ __launch_bounds__(128) void beit_attn(
    const __bf16* __restrict__ qbuf, const __bf16* __restrict__ kbuf,
    const __bf16* __restrict__ vbufT, const __bf16* __restrict__ biasbuf,
    __bf16* __restrict__ obuf) {
  __shared__ __bf16 ldsP[4][16][BK];        // per-wave P staging (4 KB)
  int wave = threadIdx.x >> 5;
  int lane = threadIdx.x & 31;
  int hi = lane >> 4, nn = lane & 15;
  int tile = blockIdx.x * 4 + wave;          // 0 .. B*NH*64-1
  int q0 = (tile & 63) * 16;
  int bh = tile >> 6;                        // b*NH + h
  int h  = bh % NH;
  int b  = bh / NH;
  const __bf16* Q  = qbuf  + (long)bh * NTOK * HD;
  const __bf16* K  = kbuf  + (long)bh * NTOK * HD;
  const __bf16* Vt = vbufT + (long)bh * HD * NTOK;  // (d, token)
  const __bf16* Bs = biasbuf + (long)h * NTOK * NTOK;

  v16bf aq0 = load_a_frag(Q + (long)q0 * HD, HD);       // d = 0..31
  v16bf aq1 = load_a_frag(Q + (long)q0 * HD + 32, HD);  // d = 32..63
  v8f o[4] = {};
  float mrow[8], lrow[8];
#pragma unroll
  for (int r = 0; r < 8; ++r) { mrow[r] = -1e30f; lrow[r] = 0.f; }

  for (int j = 0; j < NTOK; j += 32) {
    if (j + 32 < NTOK) {                     // prefetch next K/V tiles
      __builtin_prefetch(K + (long)(j + 32) * HD + lane * 32, 0, 1);
      __builtin_prefetch(Vt + (long)lane * NTOK + j + 32, 0, 1);
      __builtin_prefetch(Vt + (long)(lane + 32) * NTOK + j + 32, 0, 1);
    }
    // S = Q * K^T over 32 key columns (two 16x16 tiles)
    v8f s0 = {}, s1 = {};
    s0 = wmma_bf16(aq0, load_b_frag_nk(K + (long)j * HD, HD), s0);
    s0 = wmma_bf16(aq1, load_b_frag_nk(K + (long)j * HD + 32, HD), s0);
    s1 = wmma_bf16(aq0, load_b_frag_nk(K + (long)(j + 16) * HD, HD), s1);
    s1 = wmma_bf16(aq1, load_b_frag_nk(K + (long)(j + 16) * HD + 32, HD), s1);

    // add relative-position bias, per-row max over the 32 columns
    float rmax[8];
#pragma unroll
    for (int r = 0; r < 8; ++r) {
      int m = q0 + r + hi * 8;
      s0[r] += bf2f(Bs[(long)m * NTOK + j + nn]);
      s1[r] += bf2f(Bs[(long)m * NTOK + j + 16 + nn]);
      rmax[r] = fmaxf(s0[r], s1[r]);
    }
#pragma unroll
    for (int msk = 1; msk < 16; msk <<= 1)
#pragma unroll
      for (int r = 0; r < 8; ++r)
        rmax[r] = fmaxf(rmax[r], __shfl_xor(rmax[r], msk, 32));

    // online softmax rescale
    float alpha[8], rsum[8];
#pragma unroll
    for (int r = 0; r < 8; ++r) {
      float nm = fmaxf(mrow[r], rmax[r]);
      alpha[r] = __expf(mrow[r] - nm);
      mrow[r] = nm;
      s0[r] = __expf(s0[r] - nm);
      s1[r] = __expf(s1[r] - nm);
      rsum[r] = s0[r] + s1[r];
    }
#pragma unroll
    for (int msk = 1; msk < 16; msk <<= 1)
#pragma unroll
      for (int r = 0; r < 8; ++r)
        rsum[r] += __shfl_xor(rsum[r], msk, 32);
#pragma unroll
    for (int r = 0; r < 8; ++r) {
      lrow[r] = lrow[r] * alpha[r] + rsum[r];
#pragma unroll
      for (int t = 0; t < 4; ++t) o[t][r] *= alpha[r];
    }

    // C-layout -> A-layout transpose of P via LDS (wave-private; LDS in-order)
#pragma unroll
    for (int r = 0; r < 8; ++r) {
      int m = r + hi * 8;
      ldsP[wave][m][nn]      = f2bf(s0[r]);
      ldsP[wave][m][nn + 16] = f2bf(s1[r]);
    }
    v16bf ap = load_a_frag(&ldsP[wave][0][0], BK);
#pragma unroll
    for (int t = 0; t < 4; ++t)   // V^T rows are d -> [n][k]-major, contiguous
      o[t] = wmma_bf16(ap, load_b_frag_nk(Vt + (long)(t * 16) * NTOK + j, NTOK),
                       o[t]);
  }

  // normalize and emit in (B, N, C) bf16 layout for the projection GEMM
#pragma unroll
  for (int r = 0; r < 8; ++r) {
    float inv = 1.0f / lrow[r];
    int m = q0 + r + hi * 8;
#pragma unroll
    for (int t = 0; t < 4; ++t)
      obuf[((long)b * NTOK + m) * DIM + h * HD + t * 16 + nn] =
          f2bf(o[t][r] * inv);
  }
}

// ---------- kernel 4: projection GEMM (8192 x 768 x 768) + bias, fp32 out ----------
__global__ __launch_bounds__(256) void beit_proj_gemm(
    const __bf16* __restrict__ a, const float* __restrict__ w,
    const float* __restrict__ bias, float* __restrict__ out) {
  __shared__ unsigned ldsAw[BM][BK / 2];
  __shared__ unsigned ldsBw[BN][BK / 2];
  int m0 = blockIdx.x * BM;
  int n0 = blockIdx.y * BN;
  int wave = threadIdx.x >> 5;
  int waveM = wave >> 1, waveN = wave & 1;
  v8f acc[2][4] = {};

  for (int k0 = 0; k0 < DIM; k0 += BK) {
    __syncthreads();
    for (int i = threadIdx.x; i < (BM * BK) / 2; i += 256) {  // A already bf16
      int r = i >> 4, c = (i & 15) * 2;
      ldsAw[r][i & 15] = *(const unsigned*)(a + (long)(m0 + r) * DIM + k0 + c);
    }
    for (int i = threadIdx.x; i < (BN * BK) / 2; i += 256) {
      int r = i >> 4, c = (i & 15) * 2;
      const float* p = &w[(long)(n0 + r) * DIM + k0 + c];
      ldsBw[r][i & 15] = pack_bf2(p[0], p[1]);
    }
    if (k0 + BK < DIM) {
      __builtin_prefetch(a + (long)(m0 + (threadIdx.x & 127)) * DIM + k0 + BK, 0, 1);
      __builtin_prefetch(&w[(long)(n0 + (threadIdx.x & 127)) * DIM + k0 + BK], 0, 1);
    }
    __syncthreads();
    const __bf16* lA = (const __bf16*)ldsAw;
    const __bf16* lB = (const __bf16*)ldsBw;
    v16bf a0 = load_a_frag(lA + (waveM * 32) * BK, BK);
    v16bf a1 = load_a_frag(lA + (waveM * 32 + 16) * BK, BK);
#pragma unroll
    for (int t = 0; t < 4; ++t) {
      v16bf bf = load_b_frag_nk(lB + (waveN * 64 + t * 16) * BK, BK);
      acc[0][t] = wmma_bf16(a0, bf, acc[0][t]);
      acc[1][t] = wmma_bf16(a1, bf, acc[1][t]);
    }
  }

  int lane = threadIdx.x & 31;
  int hi = lane >> 4, nn = lane & 15;
#pragma unroll
  for (int u = 0; u < 2; ++u)
#pragma unroll
    for (int t = 0; t < 4; ++t)
#pragma unroll
      for (int r = 0; r < 8; ++r) {
        int gm = m0 + waveM * 32 + u * 16 + r + hi * 8;
        int gn = n0 + waveN * 64 + t * 16 + nn;
        out[(long)gm * DIM + gn] = acc[u][t][r] + bias[gn];
      }
}

// ---------- launcher ----------
extern "C" void kernel_launch(void* const* d_in, const int* in_sizes, int n_in,
                              void* d_out, int out_size, void* d_ws, size_t ws_size,
                              hipStream_t stream) {
  (void)in_sizes; (void)n_in; (void)out_size; (void)ws_size;
  const float* x      = (const float*)d_in[0];
  const float* qkv_w  = (const float*)d_in[1];
  const float* q_bias = (const float*)d_in[2];
  const float* v_bias = (const float*)d_in[3];
  const float* table  = (const float*)d_in[4];
  const float* proj_w = (const float*)d_in[5];
  const float* proj_b = (const float*)d_in[6];
  const int*   relidx = (const int*)d_in[7];
  float* out = (float*)d_out;

  const size_t nQKV = (size_t)BATCH * NH * NTOK * HD;       // 6,291,456 elems
  __bf16* qbuf    = (__bf16*)d_ws;
  __bf16* kbuf    = qbuf + nQKV;
  __bf16* vbufT   = kbuf + nQKV;
  __bf16* biasbuf = vbufT + nQKV;                           // NH*N*N elems
  __bf16* obuf    = biasbuf + (size_t)NH * NTOK * NTOK;     // B*N*C elems

  beit_bias_gather<<<(NH * NTOK * NTOK) / 256, 256, 0, stream>>>(table, relidx,
                                                                 biasbuf);
  beit_qkv_gemm<<<dim3((BATCH * NTOK) / BM, (3 * DIM) / BN), 256, 0, stream>>>(
      x, qkv_w, q_bias, v_bias, qbuf, kbuf, vbufT);
  beit_attn<<<(BATCH * NH * (NTOK / 16)) / 4, 128, 0, stream>>>(
      qbuf, kbuf, vbufT, biasbuf, obuf);
  beit_proj_gemm<<<dim3((BATCH * NTOK) / BM, DIM / BN), 256, 0, stream>>>(
      obuf, proj_w, proj_b, out);
}
